// SelfAttentionPooling_62672162783473
// MI455X (gfx1250) — compile-verified, matrix operand order checked by reading, not run on hardware
//
#include <hip/hip_runtime.h>

typedef __attribute__((ext_vector_type(16))) _Float16 v16h;
typedef __attribute__((ext_vector_type(8)))  float    v8f;
typedef __attribute__((ext_vector_type(4)))  float    v4f;
typedef __attribute__((ext_vector_type(2)))  float    v2f;

#define N_GRAPHS    512
#define N_PER_GRAPH 1024
#define N_NODES     (N_GRAPHS * N_PER_GRAPH)   // 524288
#define N_EDGES     4194304
#define D_FEAT      256
#define K_KEEP      512
#define N_SEL       (N_GRAPHS * K_KEEP)        // 262144

// ---------------------------------------------------------------------------
// Kernel 0: zero the two accumulator arrays (must run every launch; the
// harness does not re-poison scratch between graph replays).
// ---------------------------------------------------------------------------
__global__ __launch_bounds__(256)
void zero_accum_kernel(float* __restrict__ attn_raw, float* __restrict__ degf) {
    int i = blockIdx.x * blockDim.x + threadIdx.x;
    attn_raw[i] = 0.0f;
    degf[i]     = 0.0f;
}

// ---------------------------------------------------------------------------
// Kernel 1: support = X @ W, one wave32 per 16 rows of X.
//
// Preferred path: native fp32 WMMA (v_wmma_f32_16x16x4_f32).
//   32-bit A 16x4 layout: lanes 0-15: M=lane, {VGPR0,VGPR1}={K=k0,k0+1};
//                         lanes 16-31: M=lane-16, {K=k0+2,k0+3}.
//   B (4x16) mirrors: each lane holds its half's K-pair of W, broadcast to
//   all 16 columns -> every column of C equals support.
//   C layout: lane<16: c[j]=support[m0+j]; lane>=16: c[j]=support[m0+8+j].
// Exact f32 FMA accumulation, zero conversion VALU. Two accumulators break
// the C->C dependency chain. X loads are non-temporal (streamed once).
//
// Fallback (if the f32 builtin is not declared): f16 split-precision path.
// ---------------------------------------------------------------------------
#if __has_builtin(__builtin_amdgcn_wmma_f32_16x16x4_f32)

__global__ __launch_bounds__(256)
void support_wmma_kernel(const float* __restrict__ X,
                         const float* __restrict__ W,
                         float* __restrict__ support) {
    const int wave = (blockIdx.x * blockDim.x + threadIdx.x) >> 5;
    const int lane = threadIdx.x & 31;
    const int m0   = wave * 16;
    const int mrow = lane & 15;
    const int half = lane >> 4;                  // 0 or 1
    const float* xrow = X + (size_t)(m0 + mrow) * D_FEAT + 2 * half;
    const float* wcol = W + 2 * half;

    v8f c0 = {}, c1 = {};
#pragma unroll
    for (int k0 = 0; k0 < D_FEAT; k0 += 8) {
        v2f a0 = __builtin_nontemporal_load((const v2f*)(xrow + k0));
        v2f a1 = __builtin_nontemporal_load((const v2f*)(xrow + k0 + 4));
        v2f b0 = *(const v2f*)(wcol + k0);       // W is hot: keep temporal
        v2f b1 = *(const v2f*)(wcol + k0 + 4);
        c0 = __builtin_amdgcn_wmma_f32_16x16x4_f32(false, a0, false, b0,
                                                   (short)0, c0, false, false);
        c1 = __builtin_amdgcn_wmma_f32_16x16x4_f32(false, a1, false, b1,
                                                   (short)0, c1, false, false);
    }

    if ((lane & 15) == 0) {
#pragma unroll
        for (int j = 0; j < 8; ++j)
            support[m0 + half * 8 + j] = c0[j] + c1[j];
    }
}

#else  // fallback: f16 split-precision (hi/lo) path

__global__ __launch_bounds__(256)
void support_wmma_kernel(const float* __restrict__ X,
                         const float* __restrict__ W,
                         float* __restrict__ support) {
    const int wave = (blockIdx.x * blockDim.x + threadIdx.x) >> 5;
    const int lane = threadIdx.x & 31;
    const int m0   = wave * 16;
    const int mrow = lane & 15;
    const int half = lane >> 4;
    const int koff = half * 8;
    const float* xrow = X + (size_t)(m0 + mrow) * D_FEAT;

    v8f c = {};
#pragma unroll
    for (int k0 = 0; k0 < D_FEAT; k0 += 32) {
        float xf[16], wf[16];
#pragma unroll
        for (int i = 0; i < 8; ++i) {
            xf[i]     = xrow[k0 + koff + i];
            xf[8 + i] = xrow[k0 + koff + 16 + i];
            wf[i]     = W[k0 + koff + i];
            wf[8 + i] = W[k0 + koff + 16 + i];
        }
        v16h ahi, alo, bhi, blo;
#pragma unroll
        for (int i = 0; i < 16; ++i) {
            _Float16 h = (_Float16)xf[i];
            ahi[i] = h;
            alo[i] = (_Float16)(xf[i] - (float)h);
            _Float16 g = (_Float16)wf[i];
            bhi[i] = g;
            blo[i] = (_Float16)(wf[i] - (float)g);
        }
        c = __builtin_amdgcn_wmma_f32_16x16x32_f16(false, ahi, false, bhi,
                                                   (short)0, c, false, false);
        c = __builtin_amdgcn_wmma_f32_16x16x32_f16(false, ahi, false, blo,
                                                   (short)0, c, false, false);
        c = __builtin_amdgcn_wmma_f32_16x16x32_f16(false, alo, false, bhi,
                                                   (short)0, c, false, false);
    }
    if ((lane & 15) == 0) {
#pragma unroll
        for (int j = 0; j < 8; ++j)
            support[m0 + half * 8 + j] = c[j];
    }
}

#endif

// ---------------------------------------------------------------------------
// Kernel 2: fused edge pass — attention scatter-add and degree count.
// Edge arrays are streamed (non-temporal); support/attn/deg (2 MB each)
// stay resident in the 192 MB L2 for the random gathers/atomics.
// ---------------------------------------------------------------------------
__global__ __launch_bounds__(256)
void edge_accum_kernel(const int* __restrict__ row, const int* __restrict__ col,
                       const float* __restrict__ vals,
                       const float* __restrict__ support,
                       float* __restrict__ attn_raw, float* __restrict__ degf) {
    int e = blockIdx.x * blockDim.x + threadIdx.x;
    int r   = __builtin_nontemporal_load(row + e);
    int c   = __builtin_nontemporal_load(col + e);
    float v = __builtin_nontemporal_load(vals + e);
    atomicAdd(&attn_raw[r], v * support[c]);
    if (r != c) atomicAdd(&degf[r], 1.0f);
}

// ---------------------------------------------------------------------------
// Kernel 3: node pass — attn = tanh(raw); inv_sqrt = rsqrt(deg+1);
// diag_norm output = 1/(deg+1).
// ---------------------------------------------------------------------------
__global__ __launch_bounds__(256)
void node_finalize_kernel(float* __restrict__ attn_raw,
                          const float* __restrict__ degf,
                          float* __restrict__ inv_sqrt,
                          float* __restrict__ diag_out) {
    int i = blockIdx.x * blockDim.x + threadIdx.x;
    attn_raw[i] = tanhf(attn_raw[i]);
    float d     = degf[i] + 1.0f;
    inv_sqrt[i] = rsqrtf(d);
    __builtin_nontemporal_store(1.0f / d, diag_out + i);
}

// ---------------------------------------------------------------------------
// Kernel 4: per-graph top-k via full bitonic sort of 1024 (value,index)
// pairs in LDS. Descending by value, ascending index on ties (matches
// jax.lax.top_k ordering). One block (256 threads) per graph.
// ---------------------------------------------------------------------------
__global__ __launch_bounds__(256)
void topk_kernel(const float* __restrict__ attn,
                 int* __restrict__ node_idx,
                 float* __restrict__ mask_out) {
    __shared__ float sval[N_PER_GRAPH];
    __shared__ int   sidx[N_PER_GRAPH];
    const int g = blockIdx.x;

    for (int i = threadIdx.x; i < N_PER_GRAPH; i += 256) {
        sval[i] = attn[g * N_PER_GRAPH + i];
        sidx[i] = i;
    }
    __syncthreads();

    for (int k = 2; k <= N_PER_GRAPH; k <<= 1) {
        for (int j = k >> 1; j > 0; j >>= 1) {
            for (int t = threadIdx.x; t < N_PER_GRAPH; t += 256) {
                int ixj = t ^ j;
                if (ixj > t) {
                    bool descending = ((t & k) == 0);
                    float v0 = sval[t], v1 = sval[ixj];
                    int   i0 = sidx[t], i1 = sidx[ixj];
                    bool inOrder = (v0 > v1) || (v0 == v1 && i0 < i1);
                    if (inOrder != descending) {
                        sval[t] = v1; sval[ixj] = v0;
                        sidx[t] = i1; sidx[ixj] = i0;
                    }
                }
            }
            __syncthreads();
        }
    }

    for (int t = threadIdx.x; t < K_KEEP; t += 256) {
        node_idx[g * K_KEEP + t] = g * N_PER_GRAPH + sidx[t];
        mask_out[g * K_KEEP + t] = (float)g;   // graph_indicator[node] == g
    }
}

// ---------------------------------------------------------------------------
// Kernel 5: hidden = X[node_idx] * attn[node_idx].  One wave32 per selected
// row; each lane moves 2 x float4. X read and hidden write are both
// streaming -> non-temporal to protect L2 residency of the hot arrays.
// ---------------------------------------------------------------------------
__global__ __launch_bounds__(256)
void hidden_gather_kernel(const float* __restrict__ X,
                          const float* __restrict__ attn,
                          const int* __restrict__ node_idx,
                          float* __restrict__ hidden) {
    const int rowi = (blockIdx.x * blockDim.x + threadIdx.x) >> 5;
    const int lane = threadIdx.x & 31;
    const int nid  = node_idx[rowi];
    const float a  = attn[nid];

    const v4f* src = (const v4f*)(X + (size_t)nid * D_FEAT);
    v4f*       dst = (v4f*)(hidden + (size_t)rowi * D_FEAT);

    v4f v0 = __builtin_nontemporal_load(src + lane);
    v4f v1 = __builtin_nontemporal_load(src + lane + 32);
    v0 *= a;
    v1 *= a;
    __builtin_nontemporal_store(v0, dst + lane);
    __builtin_nontemporal_store(v1, dst + lane + 32);
}

// ---------------------------------------------------------------------------
// Kernel 6: edge_norm = (row!=col) * inv_sqrt[row] * inv_sqrt[col].
// ---------------------------------------------------------------------------
__global__ __launch_bounds__(256)
void edge_norm_kernel(const int* __restrict__ row, const int* __restrict__ col,
                      const float* __restrict__ inv_sqrt,
                      float* __restrict__ edge_norm) {
    int e = blockIdx.x * blockDim.x + threadIdx.x;
    int r = __builtin_nontemporal_load(row + e);
    int c = __builtin_nontemporal_load(col + e);
    float en = (r != c) ? inv_sqrt[r] * inv_sqrt[c] : 0.0f;
    __builtin_nontemporal_store(en, edge_norm + e);
}

// ---------------------------------------------------------------------------
extern "C" void kernel_launch(void* const* d_in, const int* in_sizes, int n_in,
                              void* d_out, int out_size, void* d_ws, size_t ws_size,
                              hipStream_t stream) {
    const int*   edge_index = (const int*)d_in[0];       // [2, E]
    const float* adj_vals   = (const float*)d_in[1];     // [E]
    const float* X          = (const float*)d_in[2];     // [N, 256]
    const float* W          = (const float*)d_in[3];     // [256, 1]
    // d_in[4] (graph_indicator) implied by layout: node i -> i / 1024.

    const int* row = edge_index;
    const int* col = edge_index + N_EDGES;

    // Output layout (flat, return order): hidden, mask, edge_norm, diag_norm
    float* out       = (float*)d_out;
    float* hidden    = out;                                    // N_SEL * 256
    float* mask_out  = hidden + (size_t)N_SEL * D_FEAT;        // N_SEL
    float* edge_norm = mask_out + N_SEL;                       // N_EDGES
    float* diag_norm = edge_norm + N_EDGES;                    // N_NODES

    // Workspace: 4 float arrays [N] + int node_idx [N_SEL]  (~9 MB)
    float* ws       = (float*)d_ws;
    float* support  = ws;
    float* attn     = ws + (size_t)N_NODES;
    float* degf     = ws + (size_t)2 * N_NODES;
    float* inv_sqrt = ws + (size_t)3 * N_NODES;
    int*   node_idx = (int*)(ws + (size_t)4 * N_NODES);

    // 0) zero accumulators (every call — graph replay safe)
    zero_accum_kernel<<<N_NODES / 256, 256, 0, stream>>>(attn, degf);

    // 1) support = X @ W  (WMMA; 32768 waves, 8 waves/block)
    support_wmma_kernel<<<(N_NODES / 16) / 8, 256, 0, stream>>>(X, W, support);

    // 2) fused edge scatter: attention sum + degree
    edge_accum_kernel<<<N_EDGES / 256, 256, 0, stream>>>(row, col, adj_vals,
                                                         support, attn, degf);

    // 3) tanh / rsqrt / diag_norm
    node_finalize_kernel<<<N_NODES / 256, 256, 0, stream>>>(attn, degf,
                                                            inv_sqrt, diag_norm);

    // 4) per-graph top-k (one block per graph)
    topk_kernel<<<N_GRAPHS, 256, 0, stream>>>(attn, node_idx, mask_out);

    // 5) hidden gather (one wave per selected row; 8 rows/block)
    hidden_gather_kernel<<<N_SEL / 8, 256, 0, stream>>>(X, attn, node_idx, hidden);

    // 6) edge_norm
    edge_norm_kernel<<<N_EDGES / 256, 256, 0, stream>>>(row, col, inv_sqrt,
                                                        edge_norm);
}
